// Attention_68058051772627
// MI455X (gfx1250) — matrix operand order repaired and checked by reading, not compiled, not run on hardware
//
#include <hip/hip_runtime.h>

// ---------------- types ----------------
typedef __bf16 bf16;
typedef __attribute__((ext_vector_type(16))) __bf16 v16bf;
typedef __attribute__((ext_vector_type(8)))  __bf16 v8bf;
typedef __attribute__((ext_vector_type(4)))  __bf16 v4bf;
typedef __attribute__((ext_vector_type(8)))  float  v8f;

// round-to-nearest-even f32 -> bf16 without relying on __bf16 arithmetic
__device__ __forceinline__ bf16 f2bf(float f) {
    unsigned u = __builtin_bit_cast(unsigned, f);
    unsigned lsb = (u >> 16) & 1u;
    u += 0x7fffu + lsb;
    unsigned short h = (unsigned short)(u >> 16);
    return __builtin_bit_cast(bf16, h);
}

__device__ __forceinline__ v8f zero_v8f() {
    v8f z;
#pragma unroll
    for (int i = 0; i < 8; ++i) z[i] = 0.0f;
    return z;
}

__device__ __forceinline__ v16bf join16(v8bf lo, v8bf hi) {
    v16bf r;
#pragma unroll
    for (int e = 0; e < 8; ++e) { r[e] = lo[e]; r[e + 8] = hi[e]; }
    return r;
}

// A-matrix fragment, 16x32 bf16 (ISA 7.12.2): lane m = row, lanes 0-15 hold
// K {8h..8h+7, 16+8h..16+8h+7} with h = lane>>4 selecting the other K-half.
__device__ __forceinline__ v16bf load_fragA(const bf16* base, int ld, int m,
                                            int khalf, int k0) {
    const bf16* p = base + (size_t)m * ld + k0;
    v8bf lo = *(const v8bf*)(p + 8 * khalf);
    v8bf hi = *(const v8bf*)(p + 16 + 8 * khalf);
    return join16(lo, hi);
}

// B-matrix fragment, 32x16 bf16: lane n = column, per-lane K
// {8h..8h+7, 16+8h..16+8h+7} (storage is row-major along K, i.e. W[n][k]).
__device__ __forceinline__ v16bf load_fragB(const bf16* base, int ld, int n,
                                            int khalf, int k0) {
    const bf16* p = base + (size_t)n * ld + k0;
    v8bf lo = *(const v8bf*)(p + 8 * khalf);
    v8bf hi = *(const v8bf*)(p + 16 + 8 * khalf);
    return join16(lo, hi);
}

__device__ __forceinline__ v8f wmma_bf16(v16bf a, v16bf b, v8f c) {
    return __builtin_amdgcn_wmma_f32_16x16x32_bf16(
        false, a, false, b, (short)0, c, false, false);
}

__device__ __forceinline__ void async_load_16B(unsigned lds_off, const void* gaddr) {
    asm volatile("global_load_async_to_lds_b128 %0, %1, off"
                 :: "v"(lds_off), "v"(gaddr) : "memory");
}

// ---------------- conversion / mask prep ----------------
__global__ void k_f32_to_bf16(const float* __restrict__ s, bf16* __restrict__ d, int n) {
    int i = (blockIdx.x * blockDim.x + threadIdx.x) * 4;
    if (i + 3 < n) {
        float4 f = *(const float4*)(s + i);
        v4bf o;
        o[0] = f2bf(f.x); o[1] = f2bf(f.y); o[2] = f2bf(f.z); o[3] = f2bf(f.w);
        *(v4bf*)(d + i) = o;
    }
}

// pm[b][i] = 1.0 for i==0 (padded always-true token), else mask[b][i-1]
__global__ void k_mask(const unsigned char* __restrict__ mk, float* __restrict__ pm) {
    int idx = blockIdx.x * blockDim.x + threadIdx.x;
    if (idx < 8 * 1024) {
        int b = idx >> 10, i = idx & 1023;
        pm[idx] = (i == 0) ? 1.0f : (mk[b * 1023 + i - 1] ? 1.0f : 0.0f);
    }
}

// ---------------- GEMM core ----------------
// C = A(M x K) * B(N x K)^T. 256 threads = 8 waves stacked vertically.
// WG tile: 256(M) x 64(N). Wave tile: 32 x 64 -> 2x4 wmma accumulators.
// B K-chunks (64 cols x 64 k = 8KB) double-buffered in LDS via async DMA;
// next chunk streams (ASYNCcnt) while current chunk is consumed.
// A fragments come straight from global (no cross-wave reuse; L2-resident).
// All fragments for a k-step are hoisted into registers before the WMMA
// chain so the backend can clause the loads and stagger the counter waits.
template <bool BF16_OUT>
__device__ __forceinline__ void gemm_core(const bf16* __restrict__ A,
                                          const bf16* __restrict__ B,
                                          const float* __restrict__ bias,
                                          bf16* __restrict__ Cb,
                                          float* __restrict__ Cf,
                                          int M, int N, int K) {
    const int tid = threadIdx.x, lane = tid & 31, wv = tid >> 5;
    const int nn = lane & 15, khalf = lane >> 4;
    const int row0 = blockIdx.x * 256 + wv * 32;
    const int col0 = blockIdx.y * 64;

    __shared__ __align__(16) bf16 Bs[2][64][64];   // [buf][n][k] 16KB

    v8f acc[2][4];
#pragma unroll
    for (int i = 0; i < 2; ++i)
#pragma unroll
        for (int j = 0; j < 4; ++j) acc[i][j] = zero_v8f();

    const int nch = K / 64;

    // prologue: stage chunk 0 into buffer 0 (512 x 16B segments, 2/thread)
#pragma unroll
    for (int s = 0; s < 2; ++s) {
        int seg = tid + s * 256;
        int n = seg >> 3, part = seg & 7;
        async_load_16B((unsigned)(unsigned long long)&Bs[0][n][part * 8],
                       B + (size_t)(col0 + n) * K + part * 8);
    }

    for (int c = 0; c < nch; ++c) {
        const int buf = c & 1;
        if (c + 1 < nch) {
            // stream next chunk into the other buffer
#pragma unroll
            for (int s = 0; s < 2; ++s) {
                int seg = tid + s * 256;
                int n = seg >> 3, part = seg & 7;
                async_load_16B((unsigned)(unsigned long long)&Bs[buf ^ 1][n][part * 8],
                               B + (size_t)(col0 + n) * K + (c + 1) * 64 + part * 8);
            }
            asm volatile("s_wait_asynccnt 2" ::: "memory");  // chunk c done, c+1 in flight
        } else {
            asm volatile("s_wait_asynccnt 0" ::: "memory");
        }
        __syncthreads();

#pragma unroll
        for (int ks = 0; ks < 2; ++ks) {
            const int k0 = c * 64 + ks * 32;
            // hoist all fragments for this k-step
            v16bf a0 = load_fragA(A + (size_t)row0 * K, K, nn, khalf, k0);
            v16bf a1 = load_fragA(A + (size_t)(row0 + 16) * K, K, nn, khalf, k0);
            v16bf bfr[4];
#pragma unroll
            for (int j = 0; j < 4; ++j)
                bfr[j] = load_fragB(&Bs[buf][0][0], 64, j * 16 + nn, khalf, ks * 32);
            // dense WMMA chain
#pragma unroll
            for (int j = 0; j < 4; ++j) {
                acc[0][j] = wmma_bf16(a0, bfr[j], acc[0][j]);
                acc[1][j] = wmma_bf16(a1, bfr[j], acc[1][j]);
            }
        }
        __syncthreads();   // all waves done with buf before it is overwritten
    }

#pragma unroll
    for (int i = 0; i < 2; ++i)
#pragma unroll
        for (int j = 0; j < 4; ++j)
#pragma unroll
            for (int r = 0; r < 8; ++r) {
                int row = row0 + 16 * i + r + 8 * khalf;
                int col = col0 + 16 * j + nn;
                if (BF16_OUT)
                    Cb[(size_t)row * N + col] = f2bf(acc[i][j][r]);
                else
                    Cf[(size_t)row * N + col] = acc[i][j][r] + bias[col];
            }
}

__global__ void k_gemm_bf16(const bf16* __restrict__ A, const bf16* __restrict__ B,
                            bf16* __restrict__ C, int M, int N, int K) {
    gemm_core<true>(A, B, nullptr, C, nullptr, M, N, K);
}

__global__ void k_gemm_f32_bias(const bf16* __restrict__ A, const bf16* __restrict__ B,
                                const float* __restrict__ bias, float* __restrict__ C,
                                int M, int N, int K) {
    gemm_core<false>(A, B, bias, nullptr, C, M, N, K);
}

// ---------------- fused flash attention ----------------
// grid.x = B*H (128), grid.y = N/64 (16); 128 threads = 4 waves, 16 q-rows/wave.
// qkv layout: [B*N, 3072] bf16; q at col h*64, k at 1024+h*64, v at 2048+h*64.
__global__ void k_flash(const bf16* __restrict__ qkv, const float* __restrict__ pm,
                        bf16* __restrict__ attnout) {
    const int bh = blockIdx.x;
    const int b = bh >> 4, h = bh & 15;
    const int tid = threadIdx.x, lane = tid & 31, wv = tid >> 5;
    const int nn = lane & 15, khalf = lane >> 4;
    const int qrow0 = blockIdx.y * 64 + wv * 16;
    const float SCALE = 0.03125f;     // 1024^-0.5
    const float MASKV = -3.0e38f;

    __shared__ __align__(16) bf16 kbuf[32][64];     // 32 keys x d=64
    __shared__ __align__(16) bf16 vT[64][32];       // transposed: [d][key]
    __shared__ __align__(16) bf16 pbuf[4][16][32];  // per-wave P tile

    // Q fragments (K-dim = d = 64 -> two 32-wide fragments), loaded once
    const bf16* qA = qkv + ((size_t)(b * 1024 + qrow0)) * 3072 + h * 64;
    v16bf qf[2];
    qf[0] = load_fragA(qA, 3072, nn, khalf, 0);
    qf[1] = load_fragA(qA, 3072, nn, khalf, 32);

    float qm[8];
#pragma unroll
    for (int r = 0; r < 8; ++r)
        qm[r] = pm[b * 1024 + qrow0 + r + 8 * khalf];

    float mrun[8], lrun[8];
    v8f oacc[4];
#pragma unroll
    for (int r = 0; r < 8; ++r) { mrun[r] = -3.0e38f; lrun[r] = 0.0f; }
#pragma unroll
    for (int dt = 0; dt < 4; ++dt) oacc[dt] = zero_v8f();

    const bf16* kglob = qkv + ((size_t)b * 1024) * 3072 + 1024 + h * 64;
    const bf16* vglob = kglob + 1024;

    for (int j0 = 0; j0 < 1024; j0 += 32) {
        __syncthreads();
        // ---- async-stage K tile (32x64 bf16 = 256 x 16B segments) ----
#pragma unroll
        for (int s = 0; s < 2; ++s) {
            int seg = tid + s * 128;
            int jj = seg >> 3, part = seg & 7;
            async_load_16B((unsigned)(unsigned long long)&kbuf[jj][part * 8],
                           kglob + (size_t)(j0 + jj) * 3072 + part * 8);
        }
        // ---- stage V transposed (coalesced read, scalar ds stores) ----
        {
            int jj = tid >> 2, dpart = (tid & 3) * 16;
            const bf16* g = vglob + (size_t)(j0 + jj) * 3072 + dpart;
            v8bf a0 = *(const v8bf*)g;
            v8bf a1 = *(const v8bf*)(g + 8);
#pragma unroll
            for (int e = 0; e < 8; ++e) {
                vT[dpart + e][jj]     = a0[e];
                vT[dpart + 8 + e][jj] = a1[e];
            }
        }
        if (j0 + 32 < 1024)
            __builtin_prefetch(kglob + (size_t)(j0 + 32 + (tid & 31)) * 3072, 0, 1);
        asm volatile("s_wait_asynccnt 0" ::: "memory");
        __syncthreads();

        // ---- S = Q K^T, two 16-key tiles (hoist K fragments) ----
        v16bf kf[2][2];
#pragma unroll
        for (int kk = 0; kk < 2; ++kk) {
            kf[kk][0] = load_fragB(&kbuf[0][0], 64, nn,      khalf, kk * 32);
            kf[kk][1] = load_fragB(&kbuf[0][0], 64, 16 + nn, khalf, kk * 32);
        }
        v8f s0 = zero_v8f(), s1 = zero_v8f();
#pragma unroll
        for (int kk = 0; kk < 2; ++kk) {
            s0 = wmma_bf16(qf[kk], kf[kk][0], s0);
            s1 = wmma_bf16(qf[kk], kf[kk][1], s1);
        }

        const float km0 = pm[b * 1024 + j0 + nn];
        const float km1 = pm[b * 1024 + j0 + 16 + nn];

        // ---- online softmax (per-row; rows live in 16-lane halves) ----
#pragma unroll
        for (int r = 0; r < 8; ++r) {
            float a = (qm[r] * km0 > 0.5f) ? s0[r] * SCALE : MASKV;
            float c = (qm[r] * km1 > 0.5f) ? s1[r] * SCALE : MASKV;
            float mx = fmaxf(a, c);
#pragma unroll
            for (int m = 1; m < 16; m <<= 1) mx = fmaxf(mx, __shfl_xor(mx, m, 32));
            float newm = fmaxf(mrun[r], mx);
            float e0 = __expf(a - newm);
            float e1 = __expf(c - newm);
            float rs = e0 + e1;
#pragma unroll
            for (int m = 1; m < 16; m <<= 1) rs += __shfl_xor(rs, m, 32);
            float alpha = __expf(mrun[r] - newm);
            mrun[r] = newm;
            lrun[r] = lrun[r] * alpha + rs;
#pragma unroll
            for (int dt = 0; dt < 4; ++dt) oacc[dt][r] *= alpha;
            int row = r + 8 * khalf;
            pbuf[wv][row][nn]      = f2bf(e0);
            pbuf[wv][row][16 + nn] = f2bf(e1);
        }
        __syncthreads();

        // ---- O += P V (hoist P and V fragments) ----
        v16bf pf = load_fragA(&pbuf[wv][0][0], 32, nn, khalf, 0);
        v16bf vf[4];
#pragma unroll
        for (int dt = 0; dt < 4; ++dt)
            vf[dt] = load_fragB(&vT[0][0], 32, dt * 16 + nn, khalf, 0);
#pragma unroll
        for (int dt = 0; dt < 4; ++dt)
            oacc[dt] = wmma_bf16(pf, vf[dt], oacc[dt]);
    }

    // ---- normalize and store [b][i][h*64 + d] ----
#pragma unroll
    for (int r = 0; r < 8; ++r) {
        float inv = 1.0f / lrun[r];
        int i = qrow0 + r + 8 * khalf;
        size_t rowoff = ((size_t)(b * 1024 + i)) * 1024 + h * 64;
#pragma unroll
        for (int dt = 0; dt < 4; ++dt)
            attnout[rowoff + dt * 16 + nn] = f2bf(oacc[dt][r] * inv);
    }
}

// ---------------- launch ----------------
extern "C" void kernel_launch(void* const* d_in, const int* in_sizes, int n_in,
                              void* d_out, int out_size, void* d_ws, size_t ws_size,
                              hipStream_t stream) {
    (void)in_sizes; (void)n_in; (void)out_size; (void)ws_size;
    const float*         x    = (const float*)d_in[0];
    const unsigned char* mask = (const unsigned char*)d_in[1];
    const float*         Wqkv = (const float*)d_in[2];
    const float*         Wout = (const float*)d_in[3];
    const float*         bout = (const float*)d_in[4];
    float*               out  = (float*)d_out;

    const size_t M = 8192;            // B*N
    bf16* xb    = (bf16*)d_ws;                         // 8192x1024
    bf16* wqkvb = xb + M * 1024;                       // 3072x1024
    bf16* woutb = wqkvb + (size_t)3072 * 1024;         // 1024x1024
    bf16* qkvb  = woutb + (size_t)1024 * 1024;         // 8192x3072
    bf16* attnb = qkvb + M * 3072;                     // 8192x1024
    float* pm   = (float*)(attnb + M * 1024);          // 8x1024

    int n;
    n = (int)(M * 1024);
    k_f32_to_bf16<<<(n / 4 + 255) / 256, 256, 0, stream>>>(x, xb, n);
    n = 3072 * 1024;
    k_f32_to_bf16<<<(n / 4 + 255) / 256, 256, 0, stream>>>(Wqkv, wqkvb, n);
    n = 1024 * 1024;
    k_f32_to_bf16<<<(n / 4 + 255) / 256, 256, 0, stream>>>(Wout, woutb, n);
    k_mask<<<32, 256, 0, stream>>>(mask, pm);

    // M/256 = 32 blocks in x
    k_gemm_bf16<<<dim3(32, 48), 256, 0, stream>>>(xb, wqkvb, qkvb, (int)M, 3072, 1024);
    k_flash<<<dim3(128, 16), 128, 0, stream>>>(qkvb, pm, attnb);
    k_gemm_f32_bias<<<dim3(32, 16), 256, 0, stream>>>(attnb, woutb, bout, out,
                                                      (int)M, 1024, 1024);
}
